// GCNEdge_58884001628498
// MI455X (gfx1250) — matrix operand order repaired and checked by reading, not compiled, not run on hardware
//
#include <hip/hip_runtime.h>

typedef __attribute__((ext_vector_type(16))) _Float16 v16h;
typedef __attribute__((ext_vector_type(8)))  float    v8f;

#define IN_FEATS  128
#define OUT_FEATS 128

// ---------------- zero workspace (sums + counts) ----------------
__global__ void gcn_zero_f32(float* __restrict__ p, int n) {
    int i = blockIdx.x * blockDim.x + threadIdx.x;
    int stride = gridDim.x * blockDim.x;
    for (; i < n; i += stride) p[i] = 0.0f;
}

// ---------------- convert W (f32) -> Wh (f16), row-major [out][in] ----------------
__global__ void gcn_cvt_w(const float* __restrict__ W, _Float16* __restrict__ Wh, int n) {
    int i = blockIdx.x * blockDim.x + threadIdx.x;
    if (i < n) Wh[i] = (_Float16)W[i];
}

// ---------------- scatter-add: one wave per edge, lane handles 4 feats ----------------
__global__ void gcn_scatter(const float* __restrict__ edge_data,
                            const int* __restrict__ dst,
                            float* __restrict__ sums,
                            float* __restrict__ counts,
                            int n_edges) {
    int gtid = blockIdx.x * blockDim.x + threadIdx.x;
    int e    = gtid >> 5;
    int lane = gtid & 31;
    if (e >= n_edges) return;

    int d = dst[e];
    const float4 v = ((const float4*)(edge_data + (size_t)e * IN_FEATS))[lane];
    float* base = sums + (size_t)d * IN_FEATS + lane * 4;
    atomicAdd(base + 0, v.x);
    atomicAdd(base + 1, v.y);
    atomicAdd(base + 2, v.z);
    atomicAdd(base + 3, v.w);
    if (lane == 0) atomicAdd(counts + d, 1.0f);
}

// ---------------- WMMA GEMM: out = relu((sums/cnt) @ W^T + b) ----------------
// One wave owns a 16-row (node) x 128-col tile. K = 128 in 4 steps of 32.
__global__ __launch_bounds__(256) void gcn_gemm_wmma(
    const float* __restrict__ sums, const float* __restrict__ counts,
    const _Float16* __restrict__ Wh, const float* __restrict__ bias,
    float* __restrict__ out, int n_nodes, int n_tiles) {

    const int lane = threadIdx.x & 31;
    const int tile = blockIdx.x * (blockDim.x >> 5) + (threadIdx.x >> 5);
    if (tile >= n_tiles) return;                 // wave-uniform: EXEC all-1s for WMMA

    const int row_base = tile * 16;
    const int half = lane >> 4;                  // 0: lanes 0-15, 1: lanes 16-31
    const int lrow = lane & 15;

    int arow = row_base + lrow;
    if (arow >= n_nodes) arow = n_nodes - 1;     // clamp loads; stores guarded below
    const float cnt  = counts[arow];
    const float invc = 1.0f / fmaxf(cnt, 1.0f);  // fold mean into A fragment

    v8f zero = {};
    v8f acc[8];
#pragma unroll
    for (int t = 0; t < 8; ++t) acc[t] = zero;

    const float* arowp = sums + (size_t)arow * IN_FEATS;

#pragma unroll
    for (int kk = 0; kk < 4; ++kk) {
        const int K0 = kk * 32;
        // A fragment, contiguous k order used here (same as B): lane<16 -> K = 0..15; lane>=16 -> K = 16..31
        const float* ap = arowp + K0 + half * 16;
        const float4 f0 = *(const float4*)(ap + 0);
        const float4 f1 = *(const float4*)(ap + 4);
        const float4 f2 = *(const float4*)(ap + 8);
        const float4 f3 = *(const float4*)(ap + 12);
        v16h a;
        a[0]  = (_Float16)(f0.x * invc); a[1]  = (_Float16)(f0.y * invc);
        a[2]  = (_Float16)(f0.z * invc); a[3]  = (_Float16)(f0.w * invc);
        a[4]  = (_Float16)(f1.x * invc); a[5]  = (_Float16)(f1.y * invc);
        a[6]  = (_Float16)(f1.z * invc); a[7]  = (_Float16)(f1.w * invc);
        a[8]  = (_Float16)(f2.x * invc); a[9]  = (_Float16)(f2.y * invc);
        a[10] = (_Float16)(f2.z * invc); a[11] = (_Float16)(f2.w * invc);
        a[12] = (_Float16)(f3.x * invc); a[13] = (_Float16)(f3.y * invc);
        a[14] = (_Float16)(f3.z * invc); a[15] = (_Float16)(f3.w * invc);

#pragma unroll
        for (int ot = 0; ot < 8; ++ot) {
            const int ocol = ot * 16 + lrow;     // B column = output feature
            // B fragment, same contiguous k order as A: lanes 0-15 hold K=0..15, lanes 16-31 hold K=16..31
            // B[k][n] = W[n][k] -> contiguous 16 halves (32B) along W's row
            const v16h bfrag =
                *(const v16h*)(Wh + (size_t)ocol * IN_FEATS + K0 + half * 16);
            acc[ot] = __builtin_amdgcn_wmma_f32_16x16x32_f16(
                false, a, false, bfrag, (short)0, acc[ot], false, false);
        }
    }

    // C/D layout: VGPR r -> row r (lanes 0-15) / row r+8 (lanes 16-31), col = lane&15
#pragma unroll
    for (int ot = 0; ot < 8; ++ot) {
        const int ocol = ot * 16 + lrow;
        const float bv = bias[ocol];
#pragma unroll
        for (int r = 0; r < 8; ++r) {
            const int orow = row_base + half * 8 + r;
            if (orow < n_nodes)
                out[(size_t)orow * OUT_FEATS + ocol] = fmaxf(acc[ot][r] + bv, 0.0f);
        }
    }
}

extern "C" void kernel_launch(void* const* d_in, const int* in_sizes, int n_in,
                              void* d_out, int out_size, void* d_ws, size_t ws_size,
                              hipStream_t stream) {
    const float* edge_data = (const float*)d_in[0];
    const int*   dst       = (const int*)d_in[1];
    const float* W         = (const float*)d_in[2];
    const float* bias      = (const float*)d_in[3];
    float* out = (float*)d_out;

    const int n_edges = in_sizes[0] / IN_FEATS;
    const int n_nodes = out_size / OUT_FEATS;

    // workspace layout: sums [n_nodes*128] f32 | counts [n_nodes] f32 | Wh [128*128] f16
    float*    sums   = (float*)d_ws;
    float*    counts = sums + (size_t)n_nodes * IN_FEATS;
    _Float16* Wh     = (_Float16*)(counts + n_nodes);

    const int n_zero = n_nodes * IN_FEATS + n_nodes;
    gcn_zero_f32<<<2048, 256, 0, stream>>>((float*)d_ws, n_zero);

    const int wn = OUT_FEATS * IN_FEATS;
    gcn_cvt_w<<<(wn + 255) / 256, 256, 0, stream>>>(W, Wh, wn);

    // one wave (32 lanes) per edge
    const long long scat_threads = (long long)n_edges * 32;
    gcn_scatter<<<(unsigned)((scat_threads + 255) / 256), 256, 0, stream>>>(
        edge_data, dst, sums, counts, n_edges);

    const int n_tiles = (n_nodes + 15) / 16;
    gcn_gemm_wmma<<<(n_tiles + 7) / 8, 256, 0, stream>>>(
        sums, counts, Wh, bias, out, n_nodes, n_tiles);
}
